// DeepContextNet_81956565942813
// MI455X (gfx1250) — compile-verified, matrix operand order checked by reading, not run on hardware
//
#include <hip/hip_runtime.h>

#define B_     256
#define L_     256
#define H_     128
#define NH_    8
#define HD_    16
#define FF_    512
#define DEPTH_ 4
#define SEQN   259                 // 3 prefix tokens + L
#define BN     (B_ * SEQN)         // 66304 rows, divisible by 64
#define CVOC_  50001
#define KCH    128                 // K-chunk staged in LDS (32 KB of f16 weights)

typedef _Float16 v16h __attribute__((ext_vector_type(16)));
typedef _Float16 v8h  __attribute__((ext_vector_type(8)));
typedef float    v8f  __attribute__((ext_vector_type(8)));
typedef unsigned int u32x4 __attribute__((ext_vector_type(4)));
typedef int          i32x4 __attribute__((ext_vector_type(4)));
typedef int          i32x8 __attribute__((ext_vector_type(8)));

#if defined(__has_builtin)
#if __has_builtin(__builtin_amdgcn_tensor_load_to_lds)
#define HAVE_TDM 1
#endif
#endif

__device__ __forceinline__ float wave_sum(float v) {
#pragma unroll
  for (int off = 16; off > 0; off >>= 1) v += __shfl_xor(v, off, 32);
  return v;
}

__device__ __forceinline__ v8f dcn_wmma(v16h a, v16h b, v8f c) {
  return __builtin_amdgcn_wmma_f32_16x16x32_f16(false, a, false, b, (short)0, c,
                                                false, false);
}

// set bit-range [lo, lo+len) of a little-endian u32-word descriptor image
__device__ __forceinline__ void dset(unsigned* w, int lo, int len,
                                     unsigned long long val) {
  for (int i = 0; i < len; ++i) {
    int bit = lo + i;
    w[bit >> 5] |= (unsigned)((val >> i) & 1ull) << (bit & 31);
  }
}

// ---------------------------------------------------------------------------
// Transpose-convert f32 weight (K x C) -> f16 W^T (C x K): WMMA B-fragments
// then load contiguously along K exactly like A-fragments.
// ---------------------------------------------------------------------------
__global__ void dcn_tcvt(const float* __restrict__ W, _Float16* __restrict__ Wt,
                         int K, int C) {
  int i = blockIdx.x * 256 + threadIdx.x;
  if (i >= K * C) return;
  int c = i / K, k = i % K;             // output index i = c*K + k
  Wt[i] = (_Float16)W[(size_t)k * C + c];
}

// ---------------------------------------------------------------------------
// Embedding + user/item feature encode -> x (BN x 128) f32
// ---------------------------------------------------------------------------
__global__ void dcn_embed(const int* __restrict__ item_seq,
                          const int* __restrict__ action_seq,
                          const int* __restrict__ time_diff,
                          const int* __restrict__ u_vals, const int* __restrict__ u_lens,
                          const float* __restrict__ u_num,
                          const int* __restrict__ i_vals, const int* __restrict__ i_lens,
                          const float* __restrict__ i_num,
                          const float* __restrict__ item_table,
                          const float* __restrict__ act_table,
                          const float* __restrict__ time_table,
                          const float* __restrict__ cat_tables,
                          const float* __restrict__ num_W,
                          const float* __restrict__ gtok,
                          float* __restrict__ X) {
  int idx = blockIdx.x;
  int b = idx / SEQN, t = idx % SEQN, c = threadIdx.x;
  float val;
  if (t == 0) {
    val = gtok[c];
  } else if (t <= 2) {
    const int*   vals = (t == 1) ? u_vals : i_vals;
    const int*   lens = (t == 1) ? u_lens : i_lens;
    const float* nums = (t == 1) ? u_num  : i_num;
    float fsum = 0.f;
#pragma unroll
    for (int f = 0; f < 8; ++f) {
      int len = lens[b * 8 + f];
      const int* vf = vals + (b * 8 + f) * 8;
      float s = 0.f;
#pragma unroll
      for (int k = 0; k < 8; ++k) {
        int vidx = vf[k];
        float e = cat_tables[((size_t)f * CVOC_ + vidx) * H_ + c];
        s += (k < len) ? e : 0.f;
      }
      fsum += s / (float)len;
    }
#pragma unroll
    for (int f = 0; f < 4; ++f) {
      const float* nu = nums + (b * 4 + f) * 32;
      const float* W  = num_W + (size_t)f * 32 * H_ + c;
      float s = 0.f;
#pragma unroll
      for (int d = 0; d < 32; ++d) s += nu[d] * W[(size_t)d * H_];
      fsum += s;
    }
    val = fsum * (1.f / 12.f);
  } else {
    int tt = t - 3;
    val = item_table[(size_t)item_seq[b * L_ + tt] * H_ + c]
        + act_table [(size_t)action_seq[b * L_ + tt] * H_ + c]
        + time_table[(size_t)time_diff[b * L_ + tt] * H_ + c];
  }
  X[(size_t)idx * H_ + c] = val;
}

// ---------------------------------------------------------------------------
// LayerNorm: x (f32 row of 128) -> f16 (one wave32 per row, 8 rows per block)
// ---------------------------------------------------------------------------
__global__ void dcn_ln(const float* __restrict__ X, const float* __restrict__ g,
                       const float* __restrict__ bta, _Float16* __restrict__ O,
                       int nrows) {
  int wave = threadIdx.x >> 5, lane = threadIdx.x & 31;
  int row = blockIdx.x * 8 + wave;
  if (row >= nrows) return;
  const float4 xv = *(const float4*)(X + (size_t)row * H_ + lane * 4);
  float s  = xv.x + xv.y + xv.z + xv.w;
  float sq = xv.x * xv.x + xv.y * xv.y + xv.z * xv.z + xv.w * xv.w;
  s  = wave_sum(s);
  sq = wave_sum(sq);
  float mean = s * (1.f / H_);
  float var  = sq * (1.f / H_) - mean * mean;
  float rstd = rsqrtf(var + 1e-5f);
  int c = lane * 4;
  float xs[4] = {xv.x, xv.y, xv.z, xv.w};
#pragma unroll
  for (int i = 0; i < 4; ++i)
    O[(size_t)row * H_ + c + i] =
        (_Float16)((xs[i] - mean) * rstd * g[c + i] + bta[c + i]);
}

// ---------------------------------------------------------------------------
// Tile epilogue: bias (+ optional exact GELU), f16 store and/or f32 residual +=
// ---------------------------------------------------------------------------
__device__ __forceinline__ void dcn_store_tile(v8f acc, int row0, int col,
                                               float bv, int C, int act,
                                               _Float16* __restrict__ O16,
                                               float* __restrict__ Rf32) {
#pragma unroll
  for (int r = 0; r < 8; ++r) {
    float v = acc[r] + bv;
    if (act == 1) v = 0.5f * v * (1.f + erff(v * 0.70710678f));
    size_t row = (size_t)(row0 + r);
    if (O16)  O16[row * C + col] = (_Float16)v;
    if (Rf32) Rf32[row * C + col] += v;
  }
}

// ---------------------------------------------------------------------------
// WMMA f16 GEMM with TDM weight staging.
//   Y(R x C) = A(R x K) * B(K x C) + bias ; Bt = B^T (C x K) f16.
//   Workgroup: 8 waves in 2x4 grid -> 64-row x 128-col macro tile.
//   Wave: 32x32 output (2x2 v_wmma_f32_16x16x32_f16 register block).
//   Per 128-K chunk the weight tile (128 cols x 128 K = 32 KB) is DMA'd into
//   LDS by the Tensor Data Mover (one tensor_load_to_lds per chunk), waited
//   with s_wait_tensorcnt, then B-fragments come from LDS (ds_load_b128)
//   while A streams from global.
// ---------------------------------------------------------------------------
__global__ void __launch_bounds__(256)
dcn_gemm_wmma(const _Float16* __restrict__ A, const _Float16* __restrict__ Bt,
              const float* __restrict__ bias, _Float16* __restrict__ O16,
              float* __restrict__ Rf32, int R, int K, int C, int act) {
  __shared__ __align__(128) _Float16 Bsh[128 * KCH];   // [localCol][k] 32 KB
  int cBands = C >> 7;
  int band = blockIdx.x / cBands;
  int cb   = blockIdx.x % cBands;
  int wav  = threadIdx.x >> 5;
  int lane = threadIdx.x & 31;
  int wr = wav >> 2, wc = wav & 3;          // 2 x 4 wave grid
  int row0 = band * 64 + wr * 32;
  int col0 = cb * 128 + wc * 32;
  int l16 = lane & 15, half = lane >> 4;

  // A fragment base: lane holds row M=l16; elem e -> K=(e/8)*16 + half*8 + e%8
  const _Float16* arow0 = A + (size_t)(row0 + l16) * K + half * 8;
  const _Float16* arow1 = arow0 + (size_t)16 * K;
  // B fragments from LDS: lane holds col N=l16 (local), same K striping
  const _Float16* bl0 = Bsh + (size_t)(wc * 32 + l16) * KCH + half * 8;
  const _Float16* bl1 = bl0 + 16 * KCH;

  v8f z = {0.f, 0.f, 0.f, 0.f, 0.f, 0.f, 0.f, 0.f};
  v8f acc00 = z, acc01 = z, acc10 = z, acc11 = z;

  for (int kc = 0; kc < K; kc += KCH) {
    __syncthreads();                      // previous chunk fully consumed
#ifdef HAVE_TDM
    if (wav == 0) {
      // Build Tensor DMA Descriptor (D#): 2-D tile, 2-byte elements.
      unsigned g0[4] = {0u, 0u, 0u, 0u};
      unsigned g1[8] = {0u, 0u, 0u, 0u, 0u, 0u, 0u, 0u};
      unsigned long long gaddr =
          (unsigned long long)(size_t)(Bt + (size_t)(cb * 128) * K + kc);
      unsigned ldsa = (unsigned)(size_t)(&Bsh[0]);   // LDS byte offset (addr[31:0])
      dset(g0, 0, 2, 1ull);                  // count = 1 (valid descriptor)
      dset(g0, 32, 32, (unsigned long long)ldsa);    // lds_addr
      dset(g0, 64, 57, gaddr);                       // global_addr (tile start)
      dset(g0, 126, 2, 2ull);                        // type = 2 ("image")
      dset(g1, 16, 2, 1ull);                 // data_size = 2 bytes
      dset(g1, 48, 32, (unsigned long long)KCH);     // tensor_dim0
      dset(g1, 80, 32, 128ull);                      // tensor_dim1
      dset(g1, 112, 16, (unsigned long long)KCH);    // tile_dim0 (K extent)
      dset(g1, 128, 16, 128ull);                     // tile_dim1 (col extent)
      dset(g1, 160, 48, (unsigned long long)K);      // tensor_dim0_stride
      u32x4 G0 = {g0[0], g0[1], g0[2], g0[3]};
      i32x8 G1 = {(int)g1[0], (int)g1[1], (int)g1[2], (int)g1[3],
                  (int)g1[4], (int)g1[5], (int)g1[6], (int)g1[7]};
      i32x4 Z4 = {0, 0, 0, 0};
#if __clang_major__ >= 23
      i32x8 Z8 = {0, 0, 0, 0, 0, 0, 0, 0};
      __builtin_amdgcn_tensor_load_to_lds(G0, G1, Z4, Z4, Z8, 0);
#else
      __builtin_amdgcn_tensor_load_to_lds(G0, G1, Z4, Z4, 0);
#endif
      __builtin_amdgcn_s_wait_tensorcnt(0);
    }
#else
    // Fallback: cooperative vector copy of the weight tile into LDS.
    for (int i = threadIdx.x; i < 128 * (KCH / 8); i += 256) {
      int cc = i / (KCH / 8), kk = (i % (KCH / 8)) * 8;
      *(v8h*)(Bsh + (size_t)cc * KCH + kk) =
          *(const v8h*)(Bt + (size_t)(cb * 128 + cc) * K + kc + kk);
    }
#endif
    __syncthreads();

    for (int k0 = 0; k0 < KCH; k0 += 32) {
      union { v16h v; v8h h[2]; } a0, a1, b0, b1;
      const _Float16* ap0 = arow0 + kc + k0;
      const _Float16* ap1 = arow1 + kc + k0;
      a0.h[0] = *(const v8h*)(ap0);
      a0.h[1] = *(const v8h*)(ap0 + 16);
      a1.h[0] = *(const v8h*)(ap1);
      a1.h[1] = *(const v8h*)(ap1 + 16);
      b0.h[0] = *(const v8h*)(bl0 + k0);
      b0.h[1] = *(const v8h*)(bl0 + k0 + 16);
      b1.h[0] = *(const v8h*)(bl1 + k0);
      b1.h[1] = *(const v8h*)(bl1 + k0 + 16);
      __builtin_prefetch(ap0 + 64, 0, 1);
      acc00 = dcn_wmma(a0.v, b0.v, acc00);
      acc01 = dcn_wmma(a0.v, b1.v, acc01);
      acc10 = dcn_wmma(a1.v, b0.v, acc10);
      acc11 = dcn_wmma(a1.v, b1.v, acc11);
    }
  }

  int colA = col0 + l16, colB = col0 + 16 + l16;
  float bvA = bias ? bias[colA] : 0.f;
  float bvB = bias ? bias[colB] : 0.f;
  int r0 = row0 + half * 8;                 // C/D layout: vgpr r -> M = r + 8*half
  int r1 = row0 + 16 + half * 8;
  dcn_store_tile(acc00, r0, colA, bvA, C, act, O16, Rf32);
  dcn_store_tile(acc01, r0, colB, bvB, C, act, O16, Rf32);
  dcn_store_tile(acc10, r1, colA, bvA, C, act, O16, Rf32);
  dcn_store_tile(acc11, r1, colB, bvB, C, act, O16, Rf32);
}

// ---------------------------------------------------------------------------
// Attention: one block per (b, head). K/V staged in LDS, single-pass flash
// softmax per query thread. QKV layout: [q(128) | k(128) | v(128)] per token.
// ---------------------------------------------------------------------------
__global__ void dcn_attn(const _Float16* __restrict__ QKV,
                         const unsigned char* __restrict__ mask,
                         _Float16* __restrict__ O) {
  __shared__ __align__(32) _Float16 Ksh[SEQN * HD_];
  __shared__ __align__(32) _Float16 Vsh[SEQN * HD_];
  __shared__ unsigned char msk[SEQN];
  int b = blockIdx.x >> 3;
  int h = blockIdx.x & 7;
  int tid = threadIdx.x;

  for (int t = tid; t < SEQN; t += 256) {
    size_t base = ((size_t)(b * SEQN + t)) * (3 * H_) + h * HD_;
    *(v8h*)(Ksh + t * HD_)     = *(const v8h*)(QKV + base + H_);
    *(v8h*)(Ksh + t * HD_ + 8) = *(const v8h*)(QKV + base + H_ + 8);
    *(v8h*)(Vsh + t * HD_)     = *(const v8h*)(QKV + base + 2 * H_);
    *(v8h*)(Vsh + t * HD_ + 8) = *(const v8h*)(QKV + base + 2 * H_ + 8);
    msk[t] = (t < 3) ? (unsigned char)1 : mask[b * L_ + (t - 3)];
  }
  __syncthreads();

  for (int t = tid; t < SEQN; t += 256) {
    size_t base = ((size_t)(b * SEQN + t)) * (3 * H_) + h * HD_;
    float q[HD_];
#pragma unroll
    for (int d = 0; d < HD_; ++d) q[d] = (float)QKV[base + d];
    float m = -3.0e38f, s = 0.f;
    float acc[HD_];
#pragma unroll
    for (int d = 0; d < HD_; ++d) acc[d] = 0.f;
    for (int j = 0; j < SEQN; ++j) {
      if (!msk[j]) continue;
      float dot = 0.f;
#pragma unroll
      for (int d = 0; d < HD_; ++d) dot += q[d] * (float)Ksh[j * HD_ + d];
      dot *= 0.25f;                                  // 1/sqrt(16)
      float mn   = fmaxf(m, dot);
      float corr = __expf(m - mn);
      float p    = __expf(dot - mn);
      s = s * corr + p;
#pragma unroll
      for (int d = 0; d < HD_; ++d)
        acc[d] = acc[d] * corr + p * (float)Vsh[j * HD_ + d];
      m = mn;
    }
    float inv = 1.f / s;
    size_t orow = ((size_t)(b * SEQN + t)) * H_ + h * HD_;
#pragma unroll
    for (int d = 0; d < HD_; ++d) O[orow + d] = (_Float16)(acc[d] * inv);
  }
}

// ---------------------------------------------------------------------------
// Head: logits = relu(latent@bott_w+bott_b)@cls_w + cls_b ; embed = l2norm(latent)
// ---------------------------------------------------------------------------
__global__ void dcn_head(const float* __restrict__ X,
                         const float* __restrict__ bw, const float* __restrict__ bb,
                         const float* __restrict__ cw, const float* __restrict__ cb,
                         float* __restrict__ out) {
  __shared__ float lat[H_];
  __shared__ float red[H_];
  int b = blockIdx.x, c = threadIdx.x;
  lat[c] = X[((size_t)b * SEQN) * H_ + c];
  __syncthreads();
  float d = 0.f;
  for (int k = 0; k < H_; ++k) d += lat[k] * bw[(size_t)k * H_ + c];
  d = fmaxf(d + bb[c], 0.f);
  red[c] = d * cw[c];
  __syncthreads();
  for (int off = 64; off > 0; off >>= 1) {
    if (c < off) red[c] += red[c + off];
    __syncthreads();
  }
  if (c == 0) out[b] = red[0] + cb[0];
  __syncthreads();
  red[c] = lat[c] * lat[c];
  __syncthreads();
  for (int off = 64; off > 0; off >>= 1) {
    if (c < off) red[c] += red[c + off];
    __syncthreads();
  }
  float nrm = fmaxf(sqrtf(red[0]), 1e-12f);
  out[B_ + (size_t)b * H_ + c] = lat[c] / nrm;
}

// ---------------------------------------------------------------------------
extern "C" void kernel_launch(void* const* d_in, const int* in_sizes, int n_in,
                              void* d_out, int out_size, void* d_ws, size_t ws_size,
                              hipStream_t stream) {
  const int*   item_seq   = (const int*)d_in[0];
  const int*   action_seq = (const int*)d_in[1];
  const int*   time_diff  = (const int*)d_in[2];
  const unsigned char* mask = (const unsigned char*)d_in[3];
  const int*   u_cat_vals = (const int*)d_in[4];
  const int*   u_cat_lens = (const int*)d_in[5];
  const float* u_num      = (const float*)d_in[6];
  const int*   i_cat_vals = (const int*)d_in[7];
  const int*   i_cat_lens = (const int*)d_in[8];
  const float* i_num      = (const float*)d_in[9];
  const float* item_table = (const float*)d_in[10];
  const float* act_table  = (const float*)d_in[11];
  const float* time_table = (const float*)d_in[12];
  const float* cat_tables = (const float*)d_in[13];
  const float* num_W      = (const float*)d_in[14];
  const float* gtok       = (const float*)d_in[15];
  const float* ln1_g      = (const float*)d_in[16];
  const float* ln1_b      = (const float*)d_in[17];
  const float* qkv_w      = (const float*)d_in[18];
  const float* qkv_b      = (const float*)d_in[19];
  const float* out_w      = (const float*)d_in[20];
  const float* out_b      = (const float*)d_in[21];
  const float* mlp_w1     = (const float*)d_in[22];
  const float* mlp_b1     = (const float*)d_in[23];
  const float* mlp_w2     = (const float*)d_in[24];
  const float* mlp_b2     = (const float*)d_in[25];
  const float* ln2_g      = (const float*)d_in[26];
  const float* ln2_b      = (const float*)d_in[27];
  const float* bott_w     = (const float*)d_in[28];
  const float* bott_b     = (const float*)d_in[29];
  const float* cls_w      = (const float*)d_in[30];
  const float* cls_b      = (const float*)d_in[31];

  // ---- workspace carve (256B aligned) ----
  char* ws = (char*)d_ws;
  size_t off = 0;
  auto alloc = [&](size_t bytes) -> void* {
    size_t o = (off + 255) & ~(size_t)255;
    off = o + bytes;
    return (void*)(ws + o);
  };
  float*     X     = (float*)    alloc((size_t)BN * H_ * 4);
  _Float16*  H16   = (_Float16*) alloc((size_t)BN * H_ * 2);
  _Float16*  QKV16 = (_Float16*) alloc((size_t)BN * FF_ * 2);  // also MLP-mid
  _Float16*  MID16 = QKV16;                                    // disjoint lifetime
  _Float16*  ATT16 = (_Float16*) alloc((size_t)BN * H_ * 2);
  _Float16*  WT    = (_Float16*) alloc((size_t)DEPTH_ * 196608 * 2);
  (void)ws_size; (void)in_sizes; (void)n_in; (void)out_size;

  const int LSTRIDE = 196608;                   // f16 elems of weights per layer
  const int OFF_QKV = 0, OFF_OW = 49152, OFF_W1 = 49152 + 16384,
            OFF_W2 = 49152 + 16384 + 65536;

  // ---- 1) weight transpose/convert to f16 ----
  for (int l = 0; l < DEPTH_; ++l) {
    _Float16* wl = WT + (size_t)l * LSTRIDE;
    dcn_tcvt<<<(49152 + 255) / 256, 256, 0, stream>>>(
        qkv_w + (size_t)l * 49152, wl + OFF_QKV, H_, 3 * H_);
    dcn_tcvt<<<(16384 + 255) / 256, 256, 0, stream>>>(
        out_w + (size_t)l * 16384, wl + OFF_OW, H_, H_);
    dcn_tcvt<<<(65536 + 255) / 256, 256, 0, stream>>>(
        mlp_w1 + (size_t)l * 65536, wl + OFF_W1, H_, FF_);
    dcn_tcvt<<<(65536 + 255) / 256, 256, 0, stream>>>(
        mlp_w2 + (size_t)l * 65536, wl + OFF_W2, FF_, H_);
  }

  // ---- 2) embeddings / encode ----
  dcn_embed<<<BN, H_, 0, stream>>>(item_seq, action_seq, time_diff,
                                   u_cat_vals, u_cat_lens, u_num,
                                   i_cat_vals, i_cat_lens, i_num,
                                   item_table, act_table, time_table,
                                   cat_tables, num_W, gtok, X);

  // ---- 3) transformer layers ----
  const int lnBlocks = BN / 8;                              // 8288
  const int rowBands = BN / 64;                             // 1036
  auto gemmBlocks = [rowBands](int C) { return rowBands * (C >> 7); };
  for (int l = 0; l < DEPTH_; ++l) {
    _Float16* wl = WT + (size_t)l * LSTRIDE;
    dcn_ln<<<lnBlocks, 256, 0, stream>>>(X, ln1_g + l * H_, ln1_b + l * H_, H16, BN);
    dcn_gemm_wmma<<<gemmBlocks(3 * H_), 256, 0, stream>>>(
        H16, wl + OFF_QKV, qkv_b + (size_t)l * 3 * H_, QKV16, nullptr,
        BN, H_, 3 * H_, 0);
    dcn_attn<<<B_ * NH_, 256, 0, stream>>>(QKV16, mask, ATT16);
    dcn_gemm_wmma<<<gemmBlocks(H_), 256, 0, stream>>>(
        ATT16, wl + OFF_OW, out_b + (size_t)l * H_, nullptr, X,
        BN, H_, H_, 0);
    dcn_ln<<<lnBlocks, 256, 0, stream>>>(X, ln2_g + l * H_, ln2_b + l * H_, H16, BN);
    dcn_gemm_wmma<<<gemmBlocks(FF_), 256, 0, stream>>>(
        H16, wl + OFF_W1, mlp_b1 + (size_t)l * FF_, MID16, nullptr,
        BN, H_, FF_, 1);
    dcn_gemm_wmma<<<gemmBlocks(H_), 256, 0, stream>>>(
        MID16, wl + OFF_W2, mlp_b2 + (size_t)l * H_, nullptr, X,
        BN, FF_, H_, 0);
  }

  // ---- 4) head ----
  dcn_head<<<B_, H_, 0, stream>>>(X, bott_w, bott_b, cls_w, cls_b, (float*)d_out);
}
